// CharRNN_26740466385336
// MI455X (gfx1250) — matrix operand order repaired and checked by reading, not compiled, not run on hardware
//
#include <hip/hip_runtime.h>

// ---------------- CDNA5 WMMA / builtin types ----------------
typedef __attribute__((ext_vector_type(16))) __bf16       v16bf;
typedef __attribute__((ext_vector_type(8)))  float        v8f;
typedef __attribute__((ext_vector_type(4)))  unsigned int v4u;
typedef __attribute__((ext_vector_type(8)))  int          v8i;
typedef __attribute__((ext_vector_type(4)))  int          v4i;

struct Chunks { v4u lo, hi; };

// Build a 16x bf16 WMMA operand from two 16-byte LDS chunks.
static __device__ __forceinline__ v16bf frag2(const __bf16* p0, const __bf16* p1) {
    Chunks c;
    c.lo = *reinterpret_cast<const v4u*>(p0);
    c.hi = *reinterpret_cast<const v4u*>(p1);
    return __builtin_bit_cast(v16bf, c);
}

static __device__ __forceinline__ float sigm(float x) {
    return 1.0f / (1.0f + __expf(-x));
}
static __device__ __forceinline__ float ftanh(float x) {
    float e = __expf(2.0f * x);
    return (e - 1.0f) / (e + 1.0f);
}

// ---------------- problem constants ----------------
constexpr int Hn = 65;        // hidden / vocab
constexpr int Bn = 50;        // batch
constexpr int Tn = 4096;      // time steps
constexpr int FH = 4 * Hn;    // 260 = gate width
constexpr int KP = 160;       // padded K for [x|h] concat (130 -> 5 x 32)
constexpr int NP = 272;       // padded N (260 -> 17 x 16)
constexpr int MP = 64;        // padded M (50 -> 4 x 16)
constexpr int KP2 = 96;       // padded K for dense (65 -> 3 x 32)

// LDS carve offsets (all multiples of 16 bytes)
constexpr int OFF_Z    = 0;                       // float [64*272]   = 69632
constexpr int OFF_H    = OFF_Z + 69632;           // float [3*50*65]  -> 39040
constexpr int OFF_C    = OFF_H + 39040;           // float [3*50*65]  -> 39040
constexpr int OFF_BIAS = OFF_C + 39040;           // float [272]      = 1088
constexpr int OFF_WT   = OFF_BIAS + 1088;         // bf16  [272*160]  = 87040
constexpr int OFF_A    = OFF_WT + 87040;          // bf16  [64*160]   = 20480
constexpr int OFF_X    = OFF_A + 20480;           // f32 x_t double buffer
constexpr int XBUF     = 13008;                   // 50*65*4 = 13000 -> 16B pad
constexpr int SMEM_BYTES = OFF_X + 2 * XBUF;      // 282336 B (< 320 KB WGP LDS)

// --------------------------------------------------------------------
// Tensor Data Mover: async 2D tile load x[:, t, :] -> LDS (f32).
// Tile: 50 rows x 65 elems, row stride = T*H elements, data_size = 4B.
// Descriptor bitfields per CDNA5 ISA ch.8 (groups 0/1; groups 2/3 zero
// for a 2D tensor). Issued by one wave; completion via s_wait_tensorcnt.
// --------------------------------------------------------------------
static __device__ __forceinline__ void tdm_load_x(const float* gsrc, unsigned lds_off) {
    unsigned long long ga = (unsigned long long)(__SIZE_TYPE__)gsrc;
    v4u g0;
    g0.x = 1u;                                   // count=1, user descriptor
    g0.y = lds_off;                              // lds_addr (bytes)
    g0.z = (unsigned)ga;                         // global_addr[31:0]
    g0.w = (unsigned)(ga >> 32) | (2u << 30);    // global_addr[56:32] | type=2
    v8i g1 = { (int)(2u << 16),                  // data_size = 4 bytes
               (int)((unsigned)Hn << 16),        // tensor_dim0 = 65
               (int)((unsigned)Bn << 16),        // tensor_dim1 = 50
               (int)((unsigned)Hn << 16),        // tile_dim0 = 65
               Bn,                               // tile_dim1 = 50
               Tn * Hn,                          // tensor_dim0_stride = 266240
               0, 0 };
    v4i gz = {0, 0, 0, 0};
#if __clang_major__ >= 23
    v8i gz8 = {0, 0, 0, 0, 0, 0, 0, 0};
    __builtin_amdgcn_tensor_load_to_lds(g0, g1, gz, gz, gz8, 0);
#else
    __builtin_amdgcn_tensor_load_to_lds(g0, g1, gz, gz, 0);
#endif
}

// =====================================================================
// Persistent recurrent kernel: one workgroup, weights/state in LDS,
// TDM-prefetched x tiles, bf16 WMMA per step, top-layer h -> workspace.
// =====================================================================
__global__ __launch_bounds__(256, 1)
void lstm_persistent(const float* __restrict__ x,  // [B,T,H]
                     const float* __restrict__ W,  // [H,4H]
                     const float* __restrict__ U,  // [H,4H]
                     const float* __restrict__ b,  // [4H]
                     __bf16* __restrict__ scr)     // [B*T, H] bf16
{
    extern __shared__ char smem[];
    float*  Zs = reinterpret_cast<float*>(smem + OFF_Z);
    float*  Hs = reinterpret_cast<float*>(smem + OFF_H);
    float*  Cs = reinterpret_cast<float*>(smem + OFF_C);
    float*  Bs = reinterpret_cast<float*>(smem + OFF_BIAS);
    __bf16* Wt = reinterpret_cast<__bf16*>(smem + OFF_WT);  // n-major [NP][KP]
    __bf16* Ab = reinterpret_cast<__bf16*>(smem + OFF_A);   // m-major [MP][KP]

    const int tid  = threadIdx.x;
    const int lane = tid & 31;
    const int wave = tid >> 5;

    // ---- one-time staging: weights (fp32 -> bf16, transposed n-major), state ----
    for (int i = tid; i < NP * KP; i += 256) {
        int n = i / KP, k = i % KP;
        float v = 0.0f;
        if (n < FH && k < 2 * Hn)
            v = (k < Hn) ? W[k * FH + n] : U[(k - Hn) * FH + n];
        Wt[n * KP + k] = (__bf16)v;
    }
    for (int i = tid; i < MP * KP; i += 256) Ab[i] = (__bf16)0.0f;   // pad rows/cols
    for (int i = tid; i < 3 * Bn * Hn; i += 256) { Hs[i] = 0.0f; Cs[i] = 0.0f; }
    for (int i = tid; i < NP; i += 256) Bs[i] = (i < FH) ? b[i] : 0.0f;
    __syncthreads();

    // kick off async TDM load of x_0 into buffer 0
    if (wave == 0) tdm_load_x(x, OFF_X);

    // wave-uniform GEMM geometry (mt is fixed per wave)
    const int mt    = wave & 3;
    const int m     = mt * 16 + (lane & 15);
    const int ka    = (lane & 16) ? 8 : 0;    // A frag: lane-half -> K +/- 8
    const int kb    = (lane & 16) ? 16 : 0;   // unused; B frags also use ka (K +/- 8)
    const int rbase = mt * 16 + ((lane & 16) ? 8 : 0);
    const int nt0   = wave >> 2;

    for (int t = 0; t < Tn; ++t) {
        const float* Xcur = reinterpret_cast<const float*>(smem + OFF_X + (t & 1) * XBUF);
        // wait for this step's TDM tile, then make it visible to all waves
        if (wave == 0) __builtin_amdgcn_s_wait_tensorcnt((short)0);
        __syncthreads();
        // prefetch next step's x tile into the other buffer
        if (wave == 0 && (t + 1) < Tn)
            tdm_load_x(x + (size_t)(t + 1) * Hn, OFF_X + ((t + 1) & 1) * XBUF);

        for (int l = 0; l < 3; ++l) {
            // ---- phase 1: assemble A = [inp(65) | h_l(t-1)(65)] in bf16 ----
            for (int i = tid; i < Bn * 2 * Hn; i += 256) {
                int bb = i / (2 * Hn), k = i % (2 * Hn);
                float v;
                if (k < Hn) {
                    v = (l == 0) ? Xcur[bb * Hn + k]
                                 : Hs[(l - 1) * Bn * Hn + bb * Hn + k];
                } else {
                    v = Hs[l * Bn * Hn + bb * Hn + (k - Hn)];
                }
                Ab[bb * KP + k] = (__bf16)v;
            }
            __syncthreads();

            // ---- phase 2: Z = A * [W;U]  via v_wmma_f32_16x16x32_bf16 ----
            // A fragments for this wave's fixed M-tile (held in VGPRs).
            v16bf af[5];
#pragma unroll
            for (int ks = 0; ks < 5; ++ks) {
                const __bf16* pa = Ab + m * KP + ks * 32 + ka;
                af[ks] = frag2(pa, pa + 16);
            }
            // B fragments: full 5-deep preload per N-tile, double-buffered
            // across N-tiles so loads overlap the WMMA chain.
            v16bf bcur[5], bnxt[5];
            {
                const int n = nt0 * 16 + (lane & 15);
#pragma unroll
                for (int ks = 0; ks < 5; ++ks) {
                    const __bf16* pb = Wt + n * KP + ks * 32 + ka;
                    bcur[ks] = frag2(pb, pb + 16);
                }
            }
            for (int nt = nt0; nt < 17; nt += 2) {
                if (nt + 2 < 17) {
                    const int nn = (nt + 2) * 16 + (lane & 15);
#pragma unroll
                    for (int ks = 0; ks < 5; ++ks) {
                        const __bf16* pb = Wt + nn * KP + ks * 32 + ka;
                        bnxt[ks] = frag2(pb, pb + 16);
                    }
                }
                v8f acc = {};
#pragma unroll
                for (int ks = 0; ks < 5; ++ks)
                    acc = __builtin_amdgcn_wmma_f32_16x16x32_bf16(
                        false, af[ks], false, bcur[ks], (short)0, acc, false, false);
                const int n = nt * 16 + (lane & 15);
#pragma unroll
                for (int r = 0; r < 8; ++r)
                    Zs[(rbase + r) * NP + n] = acc[r];
                if (nt + 2 < 17) {
#pragma unroll
                    for (int ks = 0; ks < 5; ++ks) bcur[ks] = bnxt[ks];
                }
            }
            __syncthreads();

            // ---- phase 3: gates + state update ----
            for (int i = tid; i < Bn * Hn; i += 256) {
                int bb = i / Hn, u = i % Hn;
                const float* zr = Zs + bb * NP;
                float ig = sigm (zr[u]          + Bs[u]);
                float fg = sigm (zr[Hn + u]     + Bs[Hn + u]);
                float gg = ftanh(zr[2 * Hn + u] + Bs[2 * Hn + u]);
                float og = sigm (zr[3 * Hn + u] + Bs[3 * Hn + u]);
                float cn = fg * Cs[l * Bn * Hn + i] + ig * gg;
                Cs[l * Bn * Hn + i] = cn;
                float hn = og * ftanh(cn);
                Hs[l * Bn * Hn + i] = hn;
                if (l == 2)
                    scr[((size_t)bb * Tn + t) * Hn + u] = (__bf16)hn;
            }
            __syncthreads();
        }
    }
}

// =====================================================================
// Dense epilogue: out[b,t,:] = ys[b,t,:] @ Wd + bd  over M = B*T rows,
// fully parallel across the GPU, bf16 WMMA with f32 accumulate.
// =====================================================================
__global__ __launch_bounds__(256, 4)
void dense_out(const __bf16* __restrict__ scr,  // [B*T, H] bf16
               const float*  __restrict__ Wd,   // [H, H]
               const float*  __restrict__ bd,   // [H]
               float* __restrict__ out)         // [B*T, H] fp32
{
    __shared__ __bf16 As [128 * KP2];   // 128 rows, K padded to 96
    __shared__ __bf16 Wt2[80  * KP2];   // n-major, N padded to 80
    __shared__ float  Bds[80];

    const int tid  = threadIdx.x;
    const int lane = tid & 31;
    const int wave = tid >> 5;
    const size_t R0 = (size_t)blockIdx.x * 128;

    for (int i = tid; i < 80 * KP2; i += 256) {
        int n = i / KP2, k = i % KP2;
        Wt2[i] = (n < Hn && k < Hn) ? (__bf16)Wd[k * Hn + n] : (__bf16)0.0f;
    }
    for (int i = tid; i < 128 * KP2; i += 256) {
        int r = i / KP2, k = i % KP2;
        As[i] = (k < Hn) ? scr[(R0 + r) * Hn + k] : (__bf16)0.0f;
    }
    for (int i = tid; i < 80; i += 256) Bds[i] = (i < Hn) ? bd[i] : 0.0f;
    __syncthreads();

    const int m     = wave * 16 + (lane & 15);
    const int ka    = (lane & 16) ? 8 : 0;
    const int kb    = (lane & 16) ? 16 : 0;
    const int rbase = wave * 16 + ((lane & 16) ? 8 : 0);

    // A fragments for this wave's M-tile, held in VGPRs across all N-tiles.
    v16bf af[3];
#pragma unroll
    for (int ks = 0; ks < 3; ++ks) {
        const __bf16* pa = As + m * KP2 + ks * 32 + ka;
        af[ks] = frag2(pa, pa + 16);
    }

    for (int nt = 0; nt < 5; ++nt) {
        const int n = nt * 16 + (lane & 15);
        v16bf bf[3];
#pragma unroll
        for (int ks = 0; ks < 3; ++ks) {
            const __bf16* pb = Wt2 + n * KP2 + ks * 32 + ka;
            bf[ks] = frag2(pb, pb + 16);
        }
        v8f acc = {};
#pragma unroll
        for (int ks = 0; ks < 3; ++ks)
            acc = __builtin_amdgcn_wmma_f32_16x16x32_bf16(
                false, af[ks], false, bf[ks], (short)0, acc, false, false);
        if (n < Hn) {
#pragma unroll
            for (int r = 0; r < 8; ++r)
                out[(R0 + rbase + r) * (size_t)Hn + n] = acc[r] + Bds[n];
        }
    }
}

// =====================================================================
extern "C" void kernel_launch(void* const* d_in, const int* in_sizes, int n_in,
                              void* d_out, int out_size, void* d_ws, size_t ws_size,
                              hipStream_t stream) {
    const float* x  = (const float*)d_in[0];
    const float* W  = (const float*)d_in[1];
    const float* U  = (const float*)d_in[2];
    const float* b  = (const float*)d_in[3];
    const float* Wd = (const float*)d_in[4];
    const float* bd = (const float*)d_in[5];
    float*  out = (float*)d_out;
    __bf16* scr = (__bf16*)d_ws;   // B*T*H bf16 = ~26.6 MB of workspace

    lstm_persistent<<<1, 256, SMEM_BYTES, stream>>>(x, W, U, b, scr);

    const int blocks = (Bn * Tn) / 128;   // 204800 / 128 = 1600
    dense_out<<<blocks, 256, 0, stream>>>(scr, Wd, bd, out);
}
